// TimeSformer_68942815035739
// MI455X (gfx1250) — matrix operand order repaired and checked by reading, not compiled, not run on hardware
//
#include <hip/hip_runtime.h>
#include <hip/hip_bf16.h>
#include <math.h>

// ---------------- constants (from reference) ----------------
#define BB 2
#define FF 9
#define FE 8
#define CC 3
#define HH 224
#define WW 224
#define PP 16
#define NPATCH 196       // 14*14
#define NTF 4
#define TT 784           // NTF*NPATCH
#define SEQ 2352         // TT + FE*NPATCH
#define DIM 512
#define HEADS 8
#define DHEAD 64
#define INNER 512
#define PATCH_DIM 768    // C*P*P
#define MAXRUN 10
#define THRESH 0.1f
#define SCALE 0.125f     // 64^-0.5

typedef __attribute__((ext_vector_type(16))) __bf16 bf16x16;
typedef __attribute__((ext_vector_type(8)))  __bf16 bf16x8;
typedef __attribute__((ext_vector_type(8)))  float  f32x8;

// ---------------- reduction helpers ----------------
__device__ __forceinline__ float block_reduce_sum(float v, float* sm) {
  int t = threadIdx.x;
  sm[t] = v; __syncthreads();
  for (int s = 128; s > 0; s >>= 1) { if (t < s) sm[t] += sm[t + s]; __syncthreads(); }
  float r = sm[0]; __syncthreads();
  return r;
}

// ---------------- generic batched WMMA GEMM ----------------
// D[row,col] = alpha * sum_k A[row,k]*B[k,col]  (+bias[col]) (+R[row,col])
// B given as (K x N) row-major when !TB, or as (N x K) row-major when TB.
// 3-level batch: z -> i2 = z%d2, i1 = (z/d2)%d1, i0 = z/(d1*d2).
struct GemmP {
  const float* A; long long sA0, sA1, sA2; int lda;
  const float* B; long long sB0, sB1, sB2; int ldb;
  float*       D; long long sD0, sD1, sD2; int ldd;
  const float* R; long long sR0, sR1, sR2;
  const float* bias;
  int M, N, K;
  int d1, d2;
  float alpha;
};

__device__ __forceinline__ bf16x8 cvt8(const float* src) {
  const float4* s4 = reinterpret_cast<const float4*>(src);
  float4 u = s4[0], w = s4[1];
  bf16x8 tv;
  tv[0] = (__bf16)u.x; tv[1] = (__bf16)u.y; tv[2] = (__bf16)u.z; tv[3] = (__bf16)u.w;
  tv[4] = (__bf16)w.x; tv[5] = (__bf16)w.y; tv[6] = (__bf16)w.z; tv[7] = (__bf16)w.w;
  return tv;
}

// branchless guarded gather of 8 elements with row predicate + element clamp
__device__ __forceinline__ bf16x8 cvt8_guard(const float* rowp, bool rowok,
                                             int e0, int elim) {
  bf16x8 tv;
  #pragma unroll
  for (int j = 0; j < 8; ++j) {
    int e  = e0 + j;
    int ec = e < elim ? e : (elim - 1);          // clamped, always in-bounds
    float f = rowp[ec];                          // unconditional load
    tv[j] = (__bf16)((rowok && e < elim) ? f : 0.f);
  }
  return tv;
}

template<int BM, int BN, int WM, int WN, bool TB>
__global__ __launch_bounds__(32*(BM/WM)*(BN/WN))
void gemm_wmma(GemmP p) {
  constexpr int BK = 32;
  constexpr int WAVES_N = BN / WN;
  constexpr int NTHR = 32 * (BM/WM) * (BN/WN);
  constexpr int FM = WM / 16, FN = WN / 16;
  constexpr int ITA = (BM * 4) / NTHR;              // A items per thread (exact)
  constexpr int ITB = (BN * 4) / NTHR;              // B items per thread, TB path
  constexpr int ITBN = ((BN / 8) * BK) / NTHR;      // B items per thread, !TB path
  __shared__ __bf16 As[BM][BK + 8];   // row stride 40 bf16 = 80B (16B aligned)
  __shared__ __bf16 Bs[BN][BK + 8];   // stored N-major (i.e. B^T tile)

  int z  = blockIdx.z;
  int i2 = z % p.d2; int zz = z / p.d2;
  int i1 = zz % p.d1; int i0 = zz / p.d1;
  const float* A = p.A + i0 * p.sA0 + i1 * p.sA1 + i2 * p.sA2;
  const float* B = p.B + i0 * p.sB0 + i1 * p.sB1 + i2 * p.sB2;
  float*       D = p.D + i0 * p.sD0 + i1 * p.sD1 + i2 * p.sD2;
  const float* R = p.R ? (p.R + i0 * p.sR0 + i1 * p.sR1 + i2 * p.sR2) : (const float*)0;

  int m0 = blockIdx.y * BM;
  int n0 = blockIdx.x * BN;

  int lane = threadIdx.x & 31;
  int wid  = threadIdx.x >> 5;
  int wM = wid / WAVES_N, wN = wid % WAVES_N;
  int half = lane >> 4;
  int l16  = lane & 15;

  f32x8 acc[FM][FN];
  #pragma unroll
  for (int fm = 0; fm < FM; ++fm)
    #pragma unroll
    for (int fn = 0; fn < FN; ++fn)
      #pragma unroll
      for (int r = 0; r < 8; ++r) acc[fm][fn][r] = 0.f;

  // ---- tile loaders ----
  auto fastA = [&](int k0) {                 // interior rows, full K chunk
    #pragma unroll
    for (int ii = 0; ii < ITA; ++ii) {
      int it = (int)threadIdx.x + ii * NTHR;
      int r = it >> 2, kc = (it & 3) << 3;
      bf16x8 tv = cvt8(A + (long long)(m0 + r) * p.lda + k0 + kc);
      *reinterpret_cast<bf16x8*>(&As[r][kc]) = tv;
    }
  };
  auto guardA = [&](int k0) {
    #pragma unroll
    for (int ii = 0; ii < ITA; ++ii) {
      int it = (int)threadIdx.x + ii * NTHR;
      int r = it >> 2, kc = (it & 3) << 3;
      int gm = m0 + r, gk = k0 + kc;
      bf16x8 tv;
      bool rowok = gm < p.M;
      if (rowok && gk + 8 <= p.K) {
        tv = cvt8(A + (long long)gm * p.lda + gk);
      } else {
        int gmc = rowok ? gm : (p.M - 1);
        tv = cvt8_guard(A + (long long)gmc * p.lda, rowok, gk, p.K);
      }
      *reinterpret_cast<bf16x8*>(&As[r][kc]) = tv;
    }
  };
  auto fastB = [&](int k0) {
    if constexpr (TB) {
      #pragma unroll
      for (int ii = 0; ii < ITB; ++ii) {
        int it = (int)threadIdx.x + ii * NTHR;
        int n = it >> 2, kc = (it & 3) << 3;
        bf16x8 tv = cvt8(B + (long long)(n0 + n) * p.ldb + k0 + kc);
        *reinterpret_cast<bf16x8*>(&Bs[n][kc]) = tv;
      }
    } else {
      #pragma unroll
      for (int ii = 0; ii < ITBN; ++ii) {
        int it = (int)threadIdx.x + ii * NTHR;
        int c = it & 31, nb = it >> 5;
        bf16x8 tv = cvt8(B + (long long)(k0 + c) * p.ldb + n0 + nb * 8);
        #pragma unroll
        for (int j = 0; j < 8; ++j) Bs[nb * 8 + j][c] = tv[j];
      }
    }
  };
  auto guardB = [&](int k0) {
    if constexpr (TB) {
      #pragma unroll
      for (int ii = 0; ii < ITB; ++ii) {
        int it = (int)threadIdx.x + ii * NTHR;
        int n = it >> 2, kc = (it & 3) << 3;
        int gn = n0 + n, gk = k0 + kc;
        bf16x8 tv;
        bool nok = gn < p.N;
        if (nok && gk + 8 <= p.K) {
          tv = cvt8(B + (long long)gn * p.ldb + gk);
        } else {
          int gnc = nok ? gn : (p.N - 1);
          tv = cvt8_guard(B + (long long)gnc * p.ldb, nok, gk, p.K);
        }
        *reinterpret_cast<bf16x8*>(&Bs[n][kc]) = tv;
      }
    } else {
      #pragma unroll
      for (int ii = 0; ii < ITBN; ++ii) {
        int it = (int)threadIdx.x + ii * NTHR;
        int c = it & 31, nb = it >> 5;
        int gk = k0 + c, gn0 = n0 + nb * 8;
        bf16x8 tv;
        bool kok = gk < p.K;
        if (kok && gn0 + 8 <= p.N) {
          tv = cvt8(B + (long long)gk * p.ldb + gn0);
        } else {
          int gkc = kok ? gk : (p.K - 1);
          tv = cvt8_guard(B + (long long)gkc * p.ldb, kok, gn0, p.N);
        }
        #pragma unroll
        for (int j = 0; j < 8; ++j) Bs[nb * 8 + j][c] = tv[j];
      }
    }
  };
  // ---- WMMA compute on staged tiles ----
  auto compute = [&]() {
    bf16x16 af[FM]; bf16x16 bf_[FN];
    #pragma unroll
    for (int fm = 0; fm < FM; ++fm) {
      const __bf16* ap = &As[wM * WM + fm * 16 + l16][0];
      #pragma unroll
      for (int j = 0; j < 8; ++j) { af[fm][j] = ap[8 * half + j]; af[fm][8 + j] = ap[16 + 8 * half + j]; }
    }
    #pragma unroll
    for (int fn = 0; fn < FN; ++fn) {
      const __bf16* bp = &Bs[wN * WN + fn * 16 + l16][0];
      #pragma unroll
      for (int j = 0; j < 8; ++j) { bf_[fn][j] = bp[8 * half + j]; bf_[fn][8 + j] = bp[16 + 8 * half + j]; }
    }
    #pragma unroll
    for (int fm = 0; fm < FM; ++fm)
      #pragma unroll
      for (int fn = 0; fn < FN; ++fn)
        acc[fm][fn] = __builtin_amdgcn_wmma_f32_16x16x32_bf16(
            false, af[fm], false, bf_[fn], (short)0, acc[fm][fn], false, false);
  };

  const bool interiorA = (m0 + BM <= p.M);
  const bool interiorB = (n0 + BN <= p.N);
  const int kFull = (p.K / BK) * BK;

  if (interiorA && interiorB) {
    int k0 = 0;
    for (; k0 < kFull; k0 += BK) {
      if (k0 + BK < kFull) {   // prefetch next tile rows (global_prefetch_b8)
        int rr = (int)threadIdx.x & (BM - 1);
        __builtin_prefetch((const void*)(A + (long long)(m0 + rr) * p.lda + k0 + BK), 0, 3);
        int nn = (int)threadIdx.x & (BN - 1);
        if (TB) __builtin_prefetch((const void*)(B + (long long)(n0 + nn) * p.ldb + k0 + BK), 0, 3);
      }
      fastA(k0); fastB(k0);
      __syncthreads();
      compute();
      __syncthreads();
    }
    if (k0 < p.K) {
      guardA(k0); guardB(k0);
      __syncthreads();
      compute();
      __syncthreads();
    }
  } else {
    // per-operand, per-step selection: an M-edge block still uses the fast B loader
    for (int k0 = 0; k0 < p.K; k0 += BK) {
      bool kin = (k0 + BK <= p.K);
      if (interiorA && kin) fastA(k0); else guardA(k0);
      if (interiorB && kin) fastB(k0); else guardB(k0);
      __syncthreads();
      compute();
      __syncthreads();
    }
  }

  // epilogue: C/D layout -> row = 8*half + r, col = l16
  #pragma unroll
  for (int fm = 0; fm < FM; ++fm)
    #pragma unroll
    for (int fn = 0; fn < FN; ++fn)
      #pragma unroll
      for (int r = 0; r < 8; ++r) {
        int row = m0 + wM * WM + fm * 16 + half * 8 + r;
        int col = n0 + wN * WN + fn * 16 + l16;
        if (row < p.M && col < p.N) {
          float v = p.alpha * acc[fm][fn][r];
          if (p.bias) v += p.bias[col];
          long long di = (long long)row * p.ldd + col;
          if (R) v += R[di];
          D[di] = v;
        }
      }
}

// ---------------- preprocessing kernels ----------------
__global__ void diff_mags_k(const float* __restrict__ video, float* __restrict__ mags) {
  __shared__ float sm[256];
  int blk = blockIdx.x;                  // b*FE*NPATCH + fe*NPATCH + p
  int p  = blk % NPATCH; int r1 = blk / NPATCH;
  int fe = r1 % FE;      int b  = r1 / FE;
  int hp = p / 14, wp = p % 14;
  float s = 0.f;
  for (int e = threadIdx.x; e < PATCH_DIM; e += 256) {
    int c = e >> 8; int rem = e & 255; int rr = rem >> 4; int cc = rem & 15;
    long long base = (((long long)b * FF + fe) * CC + c) * (HH * WW)
                   + (hp * 16 + rr) * WW + (wp * 16 + cc);
    float d = video[base + (long long)CC * HH * WW] - video[base];
    s += fabsf(d);
  }
  s = block_reduce_sum(s, sm);
  if (threadIdx.x == 0) mags[blk] = s * (1.f / PATCH_DIM);
}

__global__ void mask_top_k(const float* __restrict__ mags, int* __restrict__ maski,
                           float* __restrict__ mask_out) {
  __shared__ float smax[256]; __shared__ int sidx[256]; __shared__ int sany[256];
  int bf = blockIdx.x;                   // 0..B*FE-1
  const float* mg = mags + bf * NPATCH;
  int t = threadIdx.x;
  float mv = -1e30f; int mi = 0; int any = 0;
  for (int p = t; p < NPATCH; p += 256) {
    float v = mg[p];
    if (v > mv) { mv = v; mi = p; }
    if (v > THRESH) any = 1;
  }
  smax[t] = mv; sidx[t] = mi; sany[t] = any; __syncthreads();
  for (int s = 128; s > 0; s >>= 1) {
    if (t < s) {
      if (smax[t + s] > smax[t] || (smax[t + s] == smax[t] && sidx[t + s] < sidx[t])) {
        smax[t] = smax[t + s]; sidx[t] = sidx[t + s];
      }
      sany[t] |= sany[t + s];
    }
    __syncthreads();
  }
  int top = sidx[0]; int nosig = !sany[0];
  for (int p = t; p < NPATCH; p += 256) {
    int m = (mg[p] > THRESH) || (nosig && p == top);
    maski[bf * NPATCH + p] = m;
    mask_out[bf * NPATCH + p] = (float)m;
  }
}

__global__ void runs_k(const int* __restrict__ maski, int* __restrict__ runs) {
  int idx = blockIdx.x * 256 + threadIdx.x;
  if (idx >= BB * NPATCH) return;
  int b = idx / NPATCH, p = idx % NPATCH;
  int r = 0;
  for (int fe = 0; fe < FE; ++fe) {
    int m = maski[(b * FE + fe) * NPATCH + p];
    r = m ? 0 : ((r + 1 > MAXRUN) ? MAXRUN : r + 1);
    runs[(b * FE + fe) * NPATCH + p] = r;
  }
}

__global__ void build_sig_k(const float* __restrict__ video, const int* __restrict__ maski,
                            float* __restrict__ sig) {
  long long idx = (long long)blockIdx.x * 256 + threadIdx.x;
  if (idx >= (long long)BB * FE * NPATCH * PATCH_DIM) return;
  int e = idx % PATCH_DIM; long long r1 = idx / PATCH_DIM;
  int p = r1 % NPATCH; r1 /= NPATCH;
  int fe = r1 % FE; int b = r1 / FE;
  float v = 0.f;
  if (maski[(b * FE + fe) * NPATCH + p]) {
    int c = e >> 8; int rem = e & 255; int rr = rem >> 4; int cc = rem & 15;
    int hp = p / 14, wp = p % 14;
    v = video[(((long long)b * FF + fe) * CC + c) * (HH * WW)
              + (hp * 16 + rr) * WW + (wp * 16 + cc)];
  }
  sig[idx] = v;
}

__global__ void assemble_x_k(float* __restrict__ x, const float* __restrict__ target,
                             const float* __restrict__ pos, const float* __restrict__ len_emb,
                             const int* __restrict__ runs) {
  long long idx = (long long)blockIdx.x * 256 + threadIdx.x;
  if (idx >= (long long)BB * SEQ * DIM) return;
  int d = idx % DIM; long long rs = idx / DIM;
  int s = rs % SEQ; int b = rs / SEQ;
  float pe = pos[s * DIM + d];
  if (s < TT) x[idx] = target[s * DIM + d] + pe;
  else {
    int j = s - TT;                       // fe*NPATCH + p
    int run = runs[b * FE * NPATCH + j];
    x[idx] += len_emb[run * DIM + d] + pe;  // GEMM already wrote sig@w_patch + b_patch
  }
}

// ---------------- LN / softmax / elementwise ----------------
__global__ void layernorm_k(const float* __restrict__ x, float* __restrict__ y,
                            const float* __restrict__ g, const float* __restrict__ b) {
  __shared__ float sm[256];
  long long row = blockIdx.x;
  const float* xr = x + row * DIM;
  float* yr = y + row * DIM;
  int t = threadIdx.x;
  float v0 = xr[t], v1 = xr[t + 256];
  float mean = block_reduce_sum(v0 + v1, sm) * (1.f / DIM);
  float d0 = v0 - mean, d1 = v1 - mean;
  float var = block_reduce_sum(d0 * d0 + d1 * d1, sm) * (1.f / DIM);
  float inv = rsqrtf(var + 1e-5f);
  yr[t]       = d0 * inv * g[t]       + b[t];
  yr[t + 256] = d1 * inv * g[t + 256] + b[t + 256];
}

__global__ void softmax_k(float* __restrict__ x, int N) {
  __shared__ float sm[256];
  long long row = blockIdx.x;
  float* p = x + row * N;
  int t = threadIdx.x;
  float m = -3.4e38f;
  for (int i = t; i < N; i += 256) m = fmaxf(m, p[i]);
  sm[t] = m; __syncthreads();
  for (int s = 128; s > 0; s >>= 1) { if (t < s) sm[t] = fmaxf(sm[t], sm[t + s]); __syncthreads(); }
  m = sm[0]; __syncthreads();
  float sum = 0.f;
  for (int i = t; i < N; i += 256) { float e = __expf(p[i] - m); p[i] = e; sum += e; }
  sum = block_reduce_sum(sum, sm);
  float inv = 1.f / sum;
  for (int i = t; i < N; i += 256) p[i] *= inv;
}

__global__ void gelu_gate_k(const float* __restrict__ h1, float* __restrict__ h) {
  long long idx = (long long)blockIdx.x * 256 + threadIdx.x;
  if (idx >= (long long)BB * SEQ * 2048) return;
  long long row = idx / 2048; int col = idx % 2048;
  float a = h1[row * 4096 + col];
  float g = h1[row * 4096 + 2048 + col];
  h[idx] = a * (0.5f * g * (1.f + erff(g * 0.70710678118654752f)));
}

__global__ void reorder_out_k(const float* __restrict__ dec, float* __restrict__ out) {
  long long idx = (long long)blockIdx.x * 256 + threadIdx.x;
  if (idx >= (long long)BB * NTF * CC * HH * WW) return;
  int w = idx % WW; long long r1 = idx / WW;
  int h = r1 % HH; r1 /= HH;
  int c = r1 % CC; r1 /= CC;
  int ntf = r1 % NTF; int b = r1 / NTF;
  int hp = h >> 4, rr = h & 15, wp = w >> 4, cc = w & 15;
  int tt = ntf * NPATCH + hp * 14 + wp;
  out[idx] = dec[((long long)(b * TT + tt)) * PATCH_DIM + c * 256 + rr * 16 + cc];
}

// ---------------- host-side launch helpers ----------------
static void run_gemm_big(hipStream_t st, const GemmP& p, int batch, bool tb) {
  dim3 g((p.N + 63) / 64, (p.M + 127) / 128, batch);
  if (tb) hipLaunchKernelGGL((gemm_wmma<128, 64, 32, 32, true>),  g, dim3(256), 0, st, p);
  else    hipLaunchKernelGGL((gemm_wmma<128, 64, 32, 32, false>), g, dim3(256), 0, st, p);
}
static void run_gemm_small(hipStream_t st, const GemmP& p, int batch, bool tb) {
  dim3 g((p.N + 63) / 64, (p.M + 15) / 16, batch);
  if (tb) hipLaunchKernelGGL((gemm_wmma<16, 64, 16, 32, true>),  g, dim3(64), 0, st, p);
  else    hipLaunchKernelGGL((gemm_wmma<16, 64, 16, 32, false>), g, dim3(64), 0, st, p);
}

extern "C" void kernel_launch(void* const* d_in, const int* in_sizes, int n_in,
                              void* d_out, int out_size, void* d_ws, size_t ws_size,
                              hipStream_t stream) {
  (void)in_sizes; (void)n_in; (void)out_size; (void)ws_size;
  const float* video    = (const float*)d_in[0];
  const float* w_patch  = (const float*)d_in[1];
  const float* b_patch  = (const float*)d_in[2];
  const float* len_emb  = (const float*)d_in[3];
  const float* pos_emb  = (const float*)d_in[4];
  const float* target   = (const float*)d_in[5];
  const float* t_ln_g   = (const float*)d_in[6];
  const float* t_ln_b   = (const float*)d_in[7];
  const float* t_wqkv   = (const float*)d_in[8];
  const float* t_wout   = (const float*)d_in[9];
  const float* t_bout   = (const float*)d_in[10];
  const float* s_ln_g   = (const float*)d_in[11];
  const float* s_ln_b   = (const float*)d_in[12];
  const float* s_wqkv   = (const float*)d_in[13];
  const float* s_wout   = (const float*)d_in[14];
  const float* s_bout   = (const float*)d_in[15];
  const float* f_ln_g   = (const float*)d_in[16];
  const float* f_ln_b   = (const float*)d_in[17];
  const float* f_w1     = (const float*)d_in[18];
  const float* f_b1     = (const float*)d_in[19];
  const float* f_w2     = (const float*)d_in[20];
  const float* f_b2     = (const float*)d_in[21];
  const float* o_ln_g   = (const float*)d_in[22];
  const float* o_ln_b   = (const float*)d_in[23];
  const float* w_dec    = (const float*)d_in[24];
  const float* b_dec    = (const float*)d_in[25];
  float* out = (float*)d_out;
  float* mask_out = out + (long long)BB * NTF * CC * HH * WW;  // mask appended after frames

  // ---- workspace layout (floats) ----
  float* ws     = (float*)d_ws;
  float* x      = ws;                         // B*SEQ*DIM        = 2,408,448
  float* xln    = x      + 2408448;           // B*SEQ*DIM
  float* qkv    = xln    + 2408448;           // B*SEQ*1536       = 7,225,344
  float* attn   = qkv    + 7225344;           // B*SEQ*DIM
  float* sig    = attn   + 2408448;           // B*FE*NPATCH*768
  float* scores = sig    + 2408448;           // max 16*784*2352  = 29,503,488 (also FFN h1)
  float* hbuf   = scores + 29503488;          // B*SEQ*2048       = 9,633,792
  float* dec    = hbuf   + 9633792;           // B*T*768          = 1,204,224
  float* mags   = dec    + 1204224;           // 3136
  int*   maski  = (int*)(mags + 3136);        // 3136
  int*   runs   = maski + 3136;               // 3136

  // ---- preprocessing ----
  diff_mags_k<<<BB * FE * NPATCH, 256, 0, stream>>>(video, mags);
  mask_top_k<<<BB * FE, 256, 0, stream>>>(mags, maski, mask_out);
  runs_k<<<(BB * NPATCH + 255) / 256, 256, 0, stream>>>(maski, runs);
  {
    long long n = (long long)BB * FE * NPATCH * PATCH_DIM;
    build_sig_k<<<(unsigned)((n + 255) / 256), 256, 0, stream>>>(video, maski, sig);
  }
  // patch embed: sig (b, 1568, 768) @ w_patch -> x rows T.. (+ b_patch)
  {
    GemmP p{}; p.A = sig; p.sA0 = 1568LL * 768; p.lda = 768;
    p.B = w_patch; p.ldb = DIM;
    p.D = x + (long long)TT * DIM; p.sD0 = (long long)SEQ * DIM; p.ldd = DIM;
    p.bias = b_patch;
    p.M = 1568; p.N = DIM; p.K = 768; p.d1 = 1; p.d2 = 1; p.alpha = 1.f;
    run_gemm_big(stream, p, BB, false);
  }
  {
    long long n = (long long)BB * SEQ * DIM;
    assemble_x_k<<<(unsigned)((n + 255) / 256), 256, 0, stream>>>(x, target, pos_emb, len_emb, runs);
  }

  // ---- one attention block (temporal if axial, else spatial) ----
  auto attn_block = [&](const float* lng, const float* lnb, const float* wqkv,
                        const float* wout, const float* bout, bool axial) {
    layernorm_k<<<BB * SEQ, 256, 0, stream>>>(x, xln, lng, lnb);
    { // qkv: (B*SEQ,512) @ (512,1536)
      GemmP p{}; p.A = xln; p.lda = DIM; p.B = wqkv; p.ldb = 3 * INNER;
      p.D = qkv; p.ldd = 3 * INNER;
      p.M = BB * SEQ; p.N = 3 * INNER; p.K = DIM; p.d1 = 1; p.d2 = 1; p.alpha = 1.f;
      run_gemm_big(stream, p, 1, false);
    }
    // cls attention: scores = SCALE * Q[:, :T] @ K^T   (batch over b,h)
    { GemmP p{};
      p.A = qkv;        p.sA0 = (long long)SEQ * 1536; p.sA2 = 64;  p.lda = 1536;
      p.B = qkv + 512;  p.sB0 = (long long)SEQ * 1536; p.sB2 = 64;  p.ldb = 1536;
      p.D = scores;     p.sD0 = 8LL * TT * SEQ; p.sD2 = (long long)TT * SEQ; p.ldd = SEQ;
      p.M = TT; p.N = SEQ; p.K = DHEAD; p.d1 = 1; p.d2 = HEADS; p.alpha = SCALE;
      run_gemm_big(stream, p, BB * HEADS, true);
    }
    softmax_k<<<BB * HEADS * TT, 256, 0, stream>>>(scores, SEQ);
    { GemmP p{};   // cls_out = att @ V   -> attn rows 0..T-1, head slice
      p.A = scores;      p.sA0 = 8LL * TT * SEQ; p.sA2 = (long long)TT * SEQ; p.lda = SEQ;
      p.B = qkv + 1024;  p.sB0 = (long long)SEQ * 1536; p.sB2 = 64; p.ldb = 1536;
      p.D = attn;        p.sD0 = (long long)SEQ * DIM;  p.sD2 = 64; p.ldd = DIM;
      p.M = TT; p.N = DHEAD; p.K = SEQ; p.d1 = 1; p.d2 = HEADS; p.alpha = 1.f;
      run_gemm_big(stream, p, BB * HEADS, false);
    }
    if (axial) {
      // temporal: per (b,h,g=patch): qg (8 x 64), kc (792 x 64) = [cls_k(784) ; kg(8)]
      GemmP t1{};
      t1.A = qkv + (long long)TT * 1536; t1.sA0 = (long long)SEQ * 1536; t1.sA1 = 64; t1.sA2 = 1536;
      t1.lda = NPATCH * 1536;
      t1.B = qkv + 512; t1.sB0 = (long long)SEQ * 1536; t1.sB1 = 64; t1.sB2 = 0; t1.ldb = 1536;
      t1.D = scores; t1.sD0 = 8LL * NPATCH * 6336; t1.sD1 = (long long)NPATCH * 6336; t1.sD2 = 6336;
      t1.ldd = 792;
      t1.M = FE; t1.N = TT; t1.K = DHEAD; t1.d1 = HEADS; t1.d2 = NPATCH; t1.alpha = SCALE;
      run_gemm_small(stream, t1, BB * HEADS * NPATCH, true);
      GemmP t2 = t1;
      t2.B = qkv + 512 + (long long)TT * 1536; t2.sB2 = 1536; t2.ldb = NPATCH * 1536;
      t2.D = scores + TT; t2.N = FE;
      run_gemm_small(stream, t2, BB * HEADS * NPATCH, true);
      softmax_k<<<BB * HEADS * NPATCH * FE, 256, 0, stream>>>(scores, 792);
      GemmP u1{};
      u1.A = scores; u1.sA0 = 8LL * NPATCH * 6336; u1.sA1 = (long long)NPATCH * 6336; u1.sA2 = 6336;
      u1.lda = 792;
      u1.B = qkv + 1024; u1.sB0 = (long long)SEQ * 1536; u1.sB1 = 64; u1.sB2 = 0; u1.ldb = 1536;
      u1.D = attn + (long long)TT * DIM; u1.sD0 = (long long)SEQ * DIM; u1.sD1 = 64; u1.sD2 = DIM;
      u1.ldd = NPATCH * DIM;   // og is transposed back: row step l -> s = T + l*196 + g
      u1.M = FE; u1.N = DHEAD; u1.K = TT; u1.d1 = HEADS; u1.d2 = NPATCH; u1.alpha = 1.f;
      run_gemm_small(stream, u1, BB * HEADS * NPATCH, false);
      GemmP u2 = u1;
      u2.A = scores + TT; u2.K = FE;
      u2.B = qkv + 1024 + (long long)TT * 1536; u2.sB2 = 1536; u2.ldb = NPATCH * 1536;
      u2.R = u2.D; u2.sR0 = u2.sD0; u2.sR1 = u2.sD1; u2.sR2 = u2.sD2;
      run_gemm_small(stream, u2, BB * HEADS * NPATCH, false);
    } else {
      // spatial: per (b,h,f): qg (196 x 64), kc (980 x 64) = [cls_k(784) ; kg(196)]
      GemmP s1{};
      s1.A = qkv + (long long)TT * 1536; s1.sA0 = (long long)SEQ * 1536; s1.sA1 = 64;
      s1.sA2 = (long long)NPATCH * 1536; s1.lda = 1536;
      s1.B = qkv + 512; s1.sB0 = (long long)SEQ * 1536; s1.sB1 = 64; s1.sB2 = 0; s1.ldb = 1536;
      s1.D = scores; s1.sD0 = 64LL * 192080; s1.sD1 = 8LL * 192080; s1.sD2 = 192080; s1.ldd = 980;
      s1.M = NPATCH; s1.N = TT; s1.K = DHEAD; s1.d1 = HEADS; s1.d2 = FE; s1.alpha = SCALE;
      run_gemm_big(stream, s1, BB * HEADS * FE, true);
      GemmP s2 = s1;
      s2.B = qkv + 512 + (long long)TT * 1536; s2.sB2 = (long long)NPATCH * 1536; s2.ldb = 1536;
      s2.D = scores + TT; s2.N = NPATCH;
      run_gemm_big(stream, s2, BB * HEADS * FE, true);
      softmax_k<<<BB * HEADS * FE * NPATCH, 256, 0, stream>>>(scores, 980);
      GemmP v1{};
      v1.A = scores; v1.sA0 = 64LL * 192080; v1.sA1 = 8LL * 192080; v1.sA2 = 192080; v1.lda = 980;
      v1.B = qkv + 1024; v1.sB0 = (long long)SEQ * 1536; v1.sB1 = 64; v1.sB2 = 0; v1.ldb = 1536;
      v1.D = attn + (long long)TT * DIM; v1.sD0 = (long long)SEQ * DIM; v1.sD1 = 64;
      v1.sD2 = (long long)NPATCH * DIM; v1.ldd = DIM;
      v1.M = NPATCH; v1.N = DHEAD; v1.K = TT; v1.d1 = HEADS; v1.d2 = FE; v1.alpha = 1.f;
      run_gemm_big(stream, v1, BB * HEADS * FE, false);
      GemmP v2 = v1;
      v2.A = scores + TT; v2.K = NPATCH;
      v2.B = qkv + 1024 + (long long)TT * 1536; v2.sB2 = (long long)NPATCH * 1536;
      v2.R = v2.D; v2.sR0 = v2.sD0; v2.sR1 = v2.sD1; v2.sR2 = v2.sD2;
      run_gemm_big(stream, v2, BB * HEADS * FE, false);
    }
    { // out projection + residual: x = attn @ wout + bout + x
      GemmP p{}; p.A = attn; p.lda = DIM; p.B = wout; p.ldb = DIM;
      p.D = x; p.ldd = DIM; p.R = x; p.bias = bout;
      p.M = BB * SEQ; p.N = DIM; p.K = INNER; p.d1 = 1; p.d2 = 1; p.alpha = 1.f;
      run_gemm_big(stream, p, 1, false);
    }
  };

  // ---- transformer layers ----
  for (int i = 0; i < 2; ++i) {
    attn_block(t_ln_g + i * DIM, t_ln_b + i * DIM, t_wqkv + (long long)i * DIM * 3 * INNER,
               t_wout + (long long)i * INNER * DIM, t_bout + i * DIM, true);
    attn_block(s_ln_g + i * DIM, s_ln_b + i * DIM, s_wqkv + (long long)i * DIM * 3 * INNER,
               s_wout + (long long)i * INNER * DIM, s_bout + i * DIM, false);
    // FFN
    layernorm_k<<<BB * SEQ, 256, 0, stream>>>(x, xln, f_ln_g + i * DIM, f_ln_b + i * DIM);
    { GemmP p{}; p.A = xln; p.lda = DIM; p.B = f_w1 + (long long)i * DIM * 4096; p.ldb = 4096;
      p.D = scores; p.ldd = 4096; p.bias = f_b1 + (long long)i * 4096;
      p.M = BB * SEQ; p.N = 4096; p.K = DIM; p.d1 = 1; p.d2 = 1; p.alpha = 1.f;
      run_gemm_big(stream, p, 1, false);
    }
    { long long n = (long long)BB * SEQ * 2048;
      gelu_gate_k<<<(unsigned)((n + 255) / 256), 256, 0, stream>>>(scores, hbuf);
    }
    { GemmP p{}; p.A = hbuf; p.lda = 2048; p.B = f_w2 + (long long)i * 2048 * DIM; p.ldb = DIM;
      p.D = x; p.ldd = DIM; p.bias = f_b2 + (long long)i * DIM; p.R = x;
      p.M = BB * SEQ; p.N = DIM; p.K = 2048; p.d1 = 1; p.d2 = 1; p.alpha = 1.f;
      run_gemm_big(stream, p, 1, false);
    }
  }

  // ---- decode ----
  layernorm_k<<<BB * SEQ, 256, 0, stream>>>(x, xln, o_ln_g, o_ln_b);
  { GemmP p{}; p.A = xln; p.sA0 = (long long)SEQ * DIM; p.lda = DIM;
    p.B = w_dec; p.ldb = PATCH_DIM;
    p.D = dec; p.sD0 = (long long)TT * PATCH_DIM; p.ldd = PATCH_DIM; p.bias = b_dec;
    p.M = TT; p.N = PATCH_DIM; p.K = DIM; p.d1 = 1; p.d2 = 1; p.alpha = 1.f;
    run_gemm_big(stream, p, BB, false);
  }
  { long long n = (long long)BB * NTF * CC * HH * WW;
    reorder_out_k<<<(unsigned)((n + 255) / 256), 256, 0, stream>>>(dec, out);
  }
}
